// MultiHeadAttentionLayer_27479200760153
// MI455X (gfx1250) — matrix operand order repaired and checked by reading, not compiled, not run on hardware
//
#include <hip/hip_runtime.h>
#include <hip/hip_bf16.h>

// ---------------------------------------------------------------------------
// MI455X (gfx1250) multi-head attention, bf16 WMMA with f32 accumulation.
// wave32; WMMA operand layouts per CDNA5 ISA 7.12.2.
// ---------------------------------------------------------------------------

typedef __bf16 bf16_t;
typedef __attribute__((ext_vector_type(16))) __bf16 v16bf;
typedef __attribute__((ext_vector_type(8)))  float  v8f;

union B16Tile {
    v16bf v;
    uint4 u[2];
    unsigned int d[8];
};

static __device__ __forceinline__ unsigned short f2bf(float f) {
    unsigned int u = __float_as_uint(f);
    unsigned int r = u + 0x7FFFu + ((u >> 16) & 1u);   // RNE
    return (unsigned short)(r >> 16);
}

static __device__ __forceinline__ v8f wmma_bf16(const B16Tile& a, const B16Tile& b, v8f c) {
    return __builtin_amdgcn_wmma_f32_16x16x32_bf16(
        /*neg_a=*/false, a.v, /*neg_b=*/false, b.v,
        /*c_mod=*/(short)0, c, /*reuse_a=*/false, /*reuse_b=*/false);
}

// ------------------------------ cast kernel --------------------------------
__global__ void cast_f32_to_bf16(const float* __restrict__ in,
                                 unsigned short* __restrict__ out, int n4) {
    int i = blockIdx.x * blockDim.x + threadIdx.x;
    if (i >= n4) return;
    float4 f = ((const float4*)in)[i];
    unsigned long long r = (unsigned long long)f2bf(f.x)
                         | ((unsigned long long)f2bf(f.y) << 16)
                         | ((unsigned long long)f2bf(f.z) << 32)
                         | ((unsigned long long)f2bf(f.w) << 48);
    ((unsigned long long*)out)[i] = r;
}

// ------------------------------ GEMM kernel --------------------------------
// C[m,n] = sum_k A[m,k] * W[n,k] + bias[n]
// A: bf16 [8192 x 1024] row-major. W: bf16 [1024 x 1024] row-major (out,in).
// Each wave computes a 16x64 tile: one A operand tile per k-step is reused
// across 4 B tiles / 4 accumulators (2.5 b128 loads per WMMA instead of 4).
// mode 0: store bf16 at [b,h,s,dk]   (Q/K head layout)
// mode 1: store bf16 at [b,h,dk,s]   (V transposed head layout)
// mode 2: store f32  at [m,n]        (final projection -> d_out)
// grid: (M/16, N/256), block 128 (4 waves; wave w owns 64-col sub-strip w).
__global__ void gemm_wmma_bf16(const unsigned short* __restrict__ A,
                               const unsigned short* __restrict__ W,
                               const float* __restrict__ bias,
                               void* __restrict__ out, int mode) {
    const int lane = threadIdx.x & 31;
    const int w    = threadIdx.x >> 5;
    const int half = lane >> 4;
    const int l    = lane & 15;
    const int m0   = blockIdx.x * 16;
    const int n0   = blockIdx.y * 256 + w * 64;

    const unsigned short* arow  = A + (size_t)(m0 + l) * 1024;
    const unsigned short* brow0 = W + (size_t)(n0 +  0 + l) * 1024 + half * 16;
    const unsigned short* brow1 = W + (size_t)(n0 + 16 + l) * 1024 + half * 16;
    const unsigned short* brow2 = W + (size_t)(n0 + 32 + l) * 1024 + half * 16;
    const unsigned short* brow3 = W + (size_t)(n0 + 48 + l) * 1024 + half * 16;

    v8f c0 = {0.f, 0.f, 0.f, 0.f, 0.f, 0.f, 0.f, 0.f};
    v8f c1 = c0, c2 = c0, c3 = c0;
    B16Tile a, b;
    for (int k0 = 0; k0 < 1024; k0 += 32) {
        // A 16x32: lane<16 -> K 0..15; lane>=16 -> K 16..31
        a.u[0] = *(const uint4*)(arow + k0 + half * 16);
        a.u[1] = *(const uint4*)(arow + k0 + half * 16 + 8);
        // B 32x16: lane n holds column n; K 0..15 (lanes<16) / 16..31 (lanes>=16)
        b.u[0] = *(const uint4*)(brow0 + k0);
        b.u[1] = *(const uint4*)(brow0 + k0 + 8);
        c0 = wmma_bf16(a, b, c0);
        b.u[0] = *(const uint4*)(brow1 + k0);
        b.u[1] = *(const uint4*)(brow1 + k0 + 8);
        c1 = wmma_bf16(a, b, c1);
        b.u[0] = *(const uint4*)(brow2 + k0);
        b.u[1] = *(const uint4*)(brow2 + k0 + 8);
        c2 = wmma_bf16(a, b, c2);
        b.u[0] = *(const uint4*)(brow3 + k0);
        b.u[1] = *(const uint4*)(brow3 + k0 + 8);
        c3 = wmma_bf16(a, b, c3);
    }

#pragma unroll
    for (int t = 0; t < 4; ++t) {
        const v8f&  c  = (t == 0) ? c0 : (t == 1) ? c1 : (t == 2) ? c2 : c3;
        const int   n  = n0 + t * 16 + l;
        const float bn = bias[n];
#pragma unroll
        for (int r = 0; r < 8; ++r) {
            const int   m   = m0 + half * 8 + r;   // C layout: M = r (+8 for hi half)
            const float val = c[r] + bn;
            if (mode == 2) {
                ((float*)out)[(size_t)m * 1024 + n] = val;
            } else {
                const int bb = m >> 10, s = m & 1023, h = n >> 6, dk = n & 63;
                const unsigned short bv = f2bf(val);
                if (mode == 0)
                    ((unsigned short*)out)[((size_t)(bb * 16 + h) * 1024 + s) * 64 + dk] = bv;
                else
                    ((unsigned short*)out)[((size_t)(bb * 16 + h) * 64 + dk) * 1024 + s] = bv;
            }
        }
    }
}

// ---------------------------- attention kernel -----------------------------
// Per (b,h): scores = Qh Kh^T / 8, mask(score==0 -> -inf), softmax, O = P Vh.
// Wave owns a 16-row Q tile. Computes S^T = K * Q^T and O^T = V^T * P^T so
// every WMMA operand load is contiguous. Online (flash) softmax; row index
// maps to lane in the C layout, so max/sum reductions are one shfl_xor(16).
// grid: (16, 16, 8) = (qtile/4, head, batch), block 128 (4 waves).
__global__ void attn_wmma(const unsigned short* __restrict__ QH,
                          const unsigned short* __restrict__ KH,
                          const unsigned short* __restrict__ VT,
                          unsigned short* __restrict__ AH) {
    const int lane = threadIdx.x & 31;
    const int w    = threadIdx.x >> 5;
    const int half = lane >> 4;
    const int l    = lane & 15;
    const int b    = blockIdx.z;
    const int h    = blockIdx.y;
    const int q0   = (blockIdx.x * 4 + w) * 16;

    const unsigned short* Q  = QH + (size_t)(b * 16 + h) * 1024 * 64;
    const unsigned short* Kp = KH + (size_t)(b * 16 + h) * 1024 * 64;
    const unsigned short* Vt = VT + (size_t)(b * 16 + h) * 64 * 1024;

    // B operand = Q^T : column n = Q row (q0+n); held in registers for the
    // whole key loop. Two tiles cover d = 0..31 and 32..63.
    B16Tile bq0, bq1;
    {
        const unsigned short* qrow = Q + (size_t)(q0 + l) * 64 + half * 8;
        bq0.u[0] = *(const uint4*)(qrow);
        bq0.u[1] = *(const uint4*)(qrow + 16);
        bq1.u[0] = *(const uint4*)(qrow + 32);
        bq1.u[1] = *(const uint4*)(qrow + 48);
    }

    float m_run = -1e30f, l_run = 0.f;
    v8f o0 = {0.f,0.f,0.f,0.f,0.f,0.f,0.f,0.f};
    v8f o1 = o0, o2 = o0, o3 = o0;

    for (int kb = 0; kb < 1024; kb += 32) {
        // ---- scores: two 16(key)x16(qrow) tiles of S^T ----
        v8f s0 = {0.f,0.f,0.f,0.f,0.f,0.f,0.f,0.f};
        v8f s1 = s0;
        B16Tile a;
        {
            const unsigned short* kr = Kp + (size_t)(kb + l) * 64;
            a.u[0] = *(const uint4*)(kr + half * 8);
            a.u[1] = *(const uint4*)(kr + 16 + half * 8);
            s0 = wmma_bf16(a, bq0, s0);
            a.u[0] = *(const uint4*)(kr + 32 + half * 8);
            a.u[1] = *(const uint4*)(kr + 48 + half * 8);
            s0 = wmma_bf16(a, bq1, s0);
        }
        {
            const unsigned short* kr = Kp + (size_t)(kb + 16 + l) * 64;
            a.u[0] = *(const uint4*)(kr + half * 8);
            a.u[1] = *(const uint4*)(kr + 16 + half * 8);
            s1 = wmma_bf16(a, bq0, s1);
            a.u[0] = *(const uint4*)(kr + 32 + half * 8);
            a.u[1] = *(const uint4*)(kr + 48 + half * 8);
            s1 = wmma_bf16(a, bq1, s1);
        }

        // scale by 1/sqrt(64) and apply the reference's exact-zero mask
        const float ninf = -__builtin_inff();
#pragma unroll
        for (int r = 0; r < 8; ++r) {
            float v0 = s0[r] * 0.125f;  s0[r] = (v0 == 0.f) ? ninf : v0;
            float v1 = s1[r] * 0.125f;  s1[r] = (v1 == 0.f) ? ninf : v1;
        }

        // ---- online softmax (per query row == per lane, halves mirrored) ----
        float pm = -1e30f;
#pragma unroll
        for (int r = 0; r < 8; ++r) { pm = fmaxf(pm, s0[r]); pm = fmaxf(pm, s1[r]); }
        pm = fmaxf(pm, __shfl_xor(pm, 16, 32));
        const float m_new = fmaxf(m_run, pm);
        const float corr  = __expf(m_run - m_new);

        float p0[8], p1[8], psum = 0.f;
#pragma unroll
        for (int r = 0; r < 8; ++r) {
            p0[r] = __expf(s0[r] - m_new); psum += p0[r];
            p1[r] = __expf(s1[r] - m_new); psum += p1[r];
        }
        psum += __shfl_xor(psum, 16, 32);
        l_run = l_run * corr + psum;
        m_run = m_new;

        // ---- pack P^T into WMMA B layout (keys 0..31 x qrows 0..15) ----
        // lanes<16 need keys {0..7,16..23} (own tile0 + own tile1),
        // lanes>=16 need keys {8..15,24..31} (own tile0 + own tile1).
        unsigned int lo[4], hi[4];
#pragma unroll
        for (int i = 0; i < 4; ++i) {
            lo[i] = (unsigned)f2bf(p0[2 * i]) | ((unsigned)f2bf(p0[2 * i + 1]) << 16);
            hi[i] = (unsigned)f2bf(p1[2 * i]) | ((unsigned)f2bf(p1[2 * i + 1]) << 16);
        }
        B16Tile bp;
#pragma unroll
        for (int i = 0; i < 4; ++i) {
            unsigned int slo = __shfl_xor(lo[i], 16, 32);
            unsigned int shi = __shfl_xor(hi[i], 16, 32);
            bp.d[i]     = lo[i];
            bp.d[4 + i] = hi[i];
        }

        // ---- rescale running O^T, then O^T += V^T * P^T (4 dim tiles) ----
#pragma unroll
        for (int r = 0; r < 8; ++r) {
            o0[r] *= corr; o1[r] *= corr; o2[r] *= corr; o3[r] *= corr;
        }
        {
            const unsigned short* vr = Vt + (size_t)(0 + l) * 1024 + kb;
            a.u[0] = *(const uint4*)(vr + half * 8);
            a.u[1] = *(const uint4*)(vr + 16 + half * 8);
            o0 = wmma_bf16(a, bp, o0);
        }
        {
            const unsigned short* vr = Vt + (size_t)(16 + l) * 1024 + kb;
            a.u[0] = *(const uint4*)(vr + half * 8);
            a.u[1] = *(const uint4*)(vr + 16 + half * 8);
            o1 = wmma_bf16(a, bp, o1);
        }
        {
            const unsigned short* vr = Vt + (size_t)(32 + l) * 1024 + kb;
            a.u[0] = *(const uint4*)(vr + half * 8);
            a.u[1] = *(const uint4*)(vr + 16 + half * 8);
            o2 = wmma_bf16(a, bp, o2);
        }
        {
            const unsigned short* vr = Vt + (size_t)(48 + l) * 1024 + kb;
            a.u[0] = *(const uint4*)(vr + half * 8);
            a.u[1] = *(const uint4*)(vr + 16 + half * 8);
            o3 = wmma_bf16(a, bp, o3);
        }
    }

    // ---- finalize: divide by row sum; write concat-head bf16 [b,s,h*64+d] ----
    const float inv = 1.0f / l_run;
    const int   s   = q0 + l;                 // C layout: N (=qrow) per lane
    unsigned short* dst = AH + ((size_t)b * 1024 + s) * 1024 + h * 64;
#pragma unroll
    for (int r = 0; r < 8; ++r) {
        dst[ 0 + half * 8 + r] = f2bf(o0[r] * inv);
        dst[16 + half * 8 + r] = f2bf(o1[r] * inv);
        dst[32 + half * 8 + r] = f2bf(o2[r] * inv);
        dst[48 + half * 8 + r] = f2bf(o3[r] * inv);
    }
}

// ------------------------------ launcher -----------------------------------
extern "C" void kernel_launch(void* const* d_in, const int* in_sizes, int n_in,
                              void* d_out, int out_size, void* d_ws, size_t ws_size,
                              hipStream_t stream) {
    const float* q  = (const float*)d_in[0];
    const float* k  = (const float*)d_in[1];
    const float* v  = (const float*)d_in[2];
    const float* Wq = (const float*)d_in[3];
    const float* bq = (const float*)d_in[4];
    const float* Wk = (const float*)d_in[5];
    const float* bk = (const float*)d_in[6];
    const float* Wv = (const float*)d_in[7];
    const float* bv = (const float*)d_in[8];
    const float* Wf = (const float*)d_in[9];
    const float* bf = (const float*)d_in[10];

    char* ws = (char*)d_ws;
    const size_t ACT_B = (size_t)8192 * 1024 * 2;   // 16 MB bf16 activations
    const size_t W_B   = (size_t)1024 * 1024 * 2;   //  2 MB bf16 weights
    unsigned short* qb  = (unsigned short*)(ws);                       // reused as AH later
    unsigned short* kb  = (unsigned short*)(ws + ACT_B);
    unsigned short* vb  = (unsigned short*)(ws + 2 * ACT_B);
    unsigned short* Wqb = (unsigned short*)(ws + 3 * ACT_B);
    unsigned short* Wkb = (unsigned short*)(ws + 3 * ACT_B + W_B);
    unsigned short* Wvb = (unsigned short*)(ws + 3 * ACT_B + 2 * W_B);
    unsigned short* Wfb = (unsigned short*)(ws + 3 * ACT_B + 3 * W_B);
    unsigned short* QH  = (unsigned short*)(ws + 3 * ACT_B + 4 * W_B);
    unsigned short* KH  = (unsigned short*)(ws + 4 * ACT_B + 4 * W_B);
    unsigned short* VT  = (unsigned short*)(ws + 5 * ACT_B + 4 * W_B);
    unsigned short* AH  = qb;    // q bf16 no longer needed once QH is built

    // 1) casts to bf16
    {
        const int nAct4 = (8 * 1024 * 1024) / 4;     // 2,097,152
        const int nW4   = (1024 * 1024) / 4;         //   262,144
        cast_f32_to_bf16<<<nAct4 / 256, 256, 0, stream>>>(q,  qb,  nAct4);
        cast_f32_to_bf16<<<nAct4 / 256, 256, 0, stream>>>(k,  kb,  nAct4);
        cast_f32_to_bf16<<<nAct4 / 256, 256, 0, stream>>>(v,  vb,  nAct4);
        cast_f32_to_bf16<<<nW4   / 256, 256, 0, stream>>>(Wq, Wqb, nW4);
        cast_f32_to_bf16<<<nW4   / 256, 256, 0, stream>>>(Wk, Wkb, nW4);
        cast_f32_to_bf16<<<nW4   / 256, 256, 0, stream>>>(Wv, Wvb, nW4);
        cast_f32_to_bf16<<<nW4   / 256, 256, 0, stream>>>(Wf, Wfb, nW4);
    }

    // 2) head projections (WMMA GEMMs), each wave owns a 16x64 tile
    dim3 ggrid(512, 4);   // M/16 x N/256
    gemm_wmma_bf16<<<ggrid, 128, 0, stream>>>(qb, Wqb, bq, (void*)QH, 0);
    gemm_wmma_bf16<<<ggrid, 128, 0, stream>>>(kb, Wkb, bk, (void*)KH, 0);
    gemm_wmma_bf16<<<ggrid, 128, 0, stream>>>(vb, Wvb, bv, (void*)VT, 1);

    // 3) attention (flash-style, all-WMMA)
    attn_wmma<<<dim3(16, 16, 8), 128, 0, stream>>>(QH, KH, VT, AH);

    // 4) output projection -> f32 d_out
    gemm_wmma_bf16<<<ggrid, 128, 0, stream>>>(AH, Wfb, bf, d_out, 2);
}
